// ReturnTrainingSampleMixture_30528627540477
// MI455X (gfx1250) — compile-verified, matrix operand order checked by reading, not run on hardware
//
#include <hip/hip_runtime.h>
#include <hip/hip_bf16.h>
#include <cmath>

// MI455X (gfx1250) wave32 WMMA pipeline, memory-bound (~1.04 GB HBM floor):
//   h = relu(x@W1+b1); logits = h@W2+b2; e = exp(logits - rowmax)
//   R = e@Y_train;  y = mask ? R / sum_mask(R) : 0   (softmax denom cancels)
// Each GEMM wave computes a full M=64 x 32-col strip (4 M-tiles x 2 N-tiles,
// 8 v_wmma_f32_16x16x32_bf16 per K-step). The streamed B matrix is staged
// 32x32-fp32 panels at a time into LDS via global_load_async_to_lds_b128
// (ASYNCcnt-tracked, full 128B cachelines, 4-deep panel ring for latency
// hiding), then fragment-read via ds_load + cvt_pk_bf16. A reads stay in L2.
// Split-K (grid.y) restores wave parallelism; partials combined by L2 kernels.

typedef __attribute__((ext_vector_type(16))) __bf16 v16bf;
typedef __attribute__((ext_vector_type(8)))  float  v8f;

#define BDIM 64
#define DDIM 4096
#define HID  1024
#define NCLS 50000

#define PSTRIDE 36          // panel row stride in floats (bank-conflict pad)
#define NPANEL  4           // async panel ring depth

__device__ __forceinline__ int kfrag_off(int r, int half) {
    // 16-bit A/B fragment K mapping (wave32, ISA §7.12.2):
    // regs 0-3: K = half*8 + 2r (+1); regs 4-7: K = 16 + half*8 + 2(r-4) (+1)
    return (r < 4) ? (half * 8 + 2 * r) : (16 + half * 8 + 2 * (r - 4));
}

// Issue one 32(K) x 32(N) fp32 B panel as 8 async b128 loads into LDS.
__device__ __forceinline__ void issue_panel(const float* __restrict__ B,
                                            int K, int N, int n0, int k0,
                                            float* bufBase, int lane, bool kguard) {
    #pragma unroll
    for (int i = 0; i < 8; ++i) {
        const int q   = i * 32 + lane;   // 16B chunk index within panel
        const int row = q >> 3;          // K row 0..31
        const int cw  = q & 7;           // 4-float column chunk 0..7
        int grow = k0 + row;
        if (kguard && grow >= K) grow = K - 1;        // clamp; zeroed at read
        int gcol = n0 + cw * 4;
        if (gcol + 4 > N) gcol = N - 4;               // clamp last strip
        const uint64_t gaddr = (uint64_t)(uintptr_t)(B + (size_t)grow * N + gcol);
        const uint32_t loff  = (uint32_t)(uintptr_t)(bufBase + row * PSTRIDE + cw * 4);
        asm volatile("global_load_async_to_lds_b128 %0, %1, off"
                     :: "v"(loff), "v"(gaddr) : "memory");
    }
}

// One wave: C_chunk = A[0:64, kb:kb+kChunk) * B[kb:kb+kChunk, n0:n0+32)
// C layout: stacked partials, chunk p at C + p*64*N. (gridDim.y==1 -> direct.)
template<bool HAS_BIAS, bool RELU, bool KGUARD>
__global__ void wmma_gemm_strip(const float* __restrict__ A,
                                const float* __restrict__ B,
                                const float* __restrict__ bias,
                                float* __restrict__ C,
                                int K, int N, int kChunk) {
    __shared__ float panel[NPANEL][32 * PSTRIDE];

    const int lane = threadIdx.x & 31;
    const int half = lane >> 4;
    const int l15  = lane & 15;
    const int n0   = blockIdx.x << 5;              // 32-col strip (2 WMMA tiles)
    const int kb   = blockIdx.y * kChunk;
    const int niter = kChunk >> 5;
    const bool t1v = (n0 + 16) < N;                // tile1 fully valid (N%16==0)
    const int c0 = n0 + l15;
    const int lc1 = t1v ? (l15 + 16) : l15;        // tile1 col within panel

    const v8f vzero = {0.f, 0.f, 0.f, 0.f, 0.f, 0.f, 0.f, 0.f};
    v8f acc[4][2];
    #pragma unroll
    for (int mt = 0; mt < 4; ++mt) { acc[mt][0] = vzero; acc[mt][1] = vzero; }

    // Prologue: fill the async ring 3 panels deep (niter >= 16 always).
    #pragma unroll
    for (int p = 0; p < NPANEL - 1; ++p)
        issue_panel(B, K, N, n0, kb + (p << 5), &panel[p][0], lane, KGUARD);

    for (int j = 0; j < niter; ++j) {
        const int k0 = kb + (j << 5);

        // Reused ring slot was ds-read in the previous iteration; drain reads first.
        asm volatile("s_wait_dscnt 0x0" ::: "memory");
        if (j + NPANEL - 1 < niter)
            issue_panel(B, K, N, n0, kb + ((j + NPANEL - 1) << 5),
                        &panel[(j + NPANEL - 1) & (NPANEL - 1)][0], lane, KGUARD);

        // Async loads complete in order: panel j has landed once at most the
        // panels issued after it remain outstanding (8 ops each).
        const int rem = niter - 1 - j;
        if      (rem >= 3) asm volatile("s_wait_asynccnt 0x18" ::: "memory");
        else if (rem == 2) asm volatile("s_wait_asynccnt 0x10" ::: "memory");
        else if (rem == 1) asm volatile("s_wait_asynccnt 0x8"  ::: "memory");
        else               asm volatile("s_wait_asynccnt 0x0"  ::: "memory");

        const float* buf = &panel[j & (NPANEL - 1)][0];

        // B fragments for both tiles from the LDS panel.
        v16bf bf0, bf1;
        #pragma unroll
        for (int r = 0; r < 8; ++r) {
            const int ko = kfrag_off(r, half);
            const int k  = k0 + ko;
            float x00 = buf[ko * PSTRIDE + l15];
            float x01 = buf[(ko + 1) * PSTRIDE + l15];
            float x10 = buf[ko * PSTRIDE + lc1];
            float x11 = buf[(ko + 1) * PSTRIDE + lc1];
            if (KGUARD) {
                if (k >= K)     { x00 = 0.f; x10 = 0.f; }
                if (k + 1 >= K) { x01 = 0.f; x11 = 0.f; }
            }
            bf0[2 * r] = (__bf16)x00; bf0[2 * r + 1] = (__bf16)x01;
            bf1[2 * r] = (__bf16)x10; bf1[2 * r + 1] = (__bf16)x11;
        }

        // 4 M-tiles reuse both B fragments -> 8 WMMAs per K-step.
        #pragma unroll
        for (int mt = 0; mt < 4; ++mt) {
            const int mrow = mt * 16 + l15;
            v16bf a;
            #pragma unroll
            for (int r = 0; r < 8; ++r) {
                const int k = k0 + kfrag_off(r, half);
                float a0 = 0.f, a1 = 0.f;
                if (!KGUARD || (k + 1) < K) {
                    const float2 av = *(const float2*)(A + (size_t)mrow * K + k);
                    a0 = av.x; a1 = av.y;
                } else if (k < K) {
                    a0 = A[(size_t)mrow * K + k];
                }
                a[2 * r] = (__bf16)a0; a[2 * r + 1] = (__bf16)a1;
            }
            acc[mt][0] = __builtin_amdgcn_wmma_f32_16x16x32_bf16(
                false, a, false, bf0, (short)0, acc[mt][0], false, false);
            acc[mt][1] = __builtin_amdgcn_wmma_f32_16x16x32_bf16(
                false, a, false, bf1, (short)0, acc[mt][1], false, false);
        }
    }

    const float bv0 = HAS_BIAS ? bias[c0] : 0.f;
    const float bv1 = (HAS_BIAS && t1v) ? bias[n0 + 16 + l15] : 0.f;
    float* Cc = C + (size_t)blockIdx.y * BDIM * N;
    #pragma unroll
    for (int mt = 0; mt < 4; ++mt) {
        #pragma unroll
        for (int r = 0; r < 8; ++r) {
            const int m = mt * 16 + half * 8 + r;   // C/D layout: VGPR r -> M=8*half+r
            float v0 = acc[mt][0][r] + bv0;
            float w1 = acc[mt][1][r] + bv1;
            if (RELU) { v0 = fmaxf(v0, 0.f); w1 = fmaxf(w1, 0.f); }
            Cc[(size_t)m * N + c0] = v0;
            if (t1v) Cc[(size_t)m * N + n0 + 16 + l15] = w1;
        }
    }
}

// h = relu(sum_p ph[p] + b1)   over 64x1024
__global__ void combine_bias_relu_kernel(const float* __restrict__ ph,
                                         const float* __restrict__ b1,
                                         float* __restrict__ h) {
    const int idx = blockIdx.x * 256 + threadIdx.x;
    if (idx >= BDIM * HID) return;
    float s = 0.f;
    #pragma unroll
    for (int p = 0; p < 8; ++p) s += ph[(size_t)p * BDIM * HID + idx];
    h[idx] = fmaxf(s + b1[idx % HID], 0.f);
}

// R = sum_p pR[p]   over 64x4096
__global__ void combine_R_kernel(const float* __restrict__ pR,
                                 float* __restrict__ R) {
    const int idx = blockIdx.x * 256 + threadIdx.x;
    if (idx >= BDIM * DDIM) return;
    float s = 0.f;
    #pragma unroll
    for (int p = 0; p < 8; ++p) s += pR[(size_t)p * BDIM * DDIM + idx];
    R[idx] = s;
}

__global__ void rowmax_kernel(const float* __restrict__ logits,
                              float* __restrict__ rmax) {
    __shared__ float sm[256];
    const int b = blockIdx.x;
    float m = -INFINITY;
    for (int i = threadIdx.x; i < NCLS; i += 256)
        m = fmaxf(m, logits[(size_t)b * NCLS + i]);
    sm[threadIdx.x] = m;
    __syncthreads();
    for (int s = 128; s > 0; s >>= 1) {
        if (threadIdx.x < s)
            sm[threadIdx.x] = fmaxf(sm[threadIdx.x], sm[threadIdx.x + s]);
        __syncthreads();
    }
    if (threadIdx.x == 0) rmax[b] = sm[0];
}

__global__ void exp_kernel(float* __restrict__ e,
                           const float* __restrict__ rmax) {
    const size_t idx = (size_t)blockIdx.x * 256 + threadIdx.x;
    if (idx < (size_t)BDIM * NCLS) {
        const int b = (int)(idx / NCLS);
        e[idx] = __expf(e[idx] - rmax[b]);
    }
}

// y[b][d] = (x!=0) ? R[b][d] / sum_{x!=0} R[b][:] : 0   (softmax denom cancels)
__global__ void masked_norm_kernel(const float* __restrict__ R,
                                   const float* __restrict__ x,
                                   float* __restrict__ y) {
    __shared__ float ssum[256];
    __shared__ int   sany[256];
    const int b = blockIdx.x;
    float s = 0.f;
    int any = 0;
    for (int i = threadIdx.x; i < DDIM; i += 256) {
        const float xv = x[(size_t)b * DDIM + i];
        if (xv != 0.f) { s += R[(size_t)b * DDIM + i]; any = 1; }
    }
    ssum[threadIdx.x] = s;
    sany[threadIdx.x] = any;
    __syncthreads();
    for (int st = 128; st > 0; st >>= 1) {
        if (threadIdx.x < st) {
            ssum[threadIdx.x] += ssum[threadIdx.x + st];
            sany[threadIdx.x] |= sany[threadIdx.x + st];
        }
        __syncthreads();
    }
    const float denom = sany[0] ? ssum[0] : 1.0f;
    for (int i = threadIdx.x; i < DDIM; i += 256) {
        const float xv = x[(size_t)b * DDIM + i];
        y[(size_t)b * DDIM + i] = (xv != 0.f) ? R[(size_t)b * DDIM + i] / denom : 0.f;
    }
}

extern "C" void kernel_launch(void* const* d_in, const int* in_sizes, int n_in,
                              void* d_out, int out_size, void* d_ws, size_t ws_size,
                              hipStream_t stream) {
    const float* x  = (const float*)d_in[0];   // (64, 4096)
    const float* W1 = (const float*)d_in[1];   // (4096, 1024)
    const float* b1 = (const float*)d_in[2];   // (1024,)
    const float* W2 = (const float*)d_in[3];   // (1024, 50000)
    const float* b2 = (const float*)d_in[4];   // (50000,)
    const float* Yt = (const float*)d_in[5];   // (50000, 4096)
    float* y = (float*)d_out;                  // (64, 4096)

    // Workspace (floats): ph[8x64x1024] | h | e | rmax | pR[8x64x4096] | R  (~24.6 MB)
    float* ph   = (float*)d_ws;
    float* h    = ph + (size_t)8 * BDIM * HID;
    float* e    = h + (size_t)BDIM * HID;
    float* rmax = e + (size_t)BDIM * NCLS;
    float* pR   = rmax + BDIM;
    float* R    = pR + (size_t)8 * BDIM * DDIM;

    const dim3 wave(32, 1, 1);

    // GEMM1 (split-K x8): ph[p] = x @ W1 over K chunk of 512  (M=64,K=4096,N=1024)
    wmma_gemm_strip<false, false, false>
        <<<dim3(HID / 32, 8), wave, 0, stream>>>(x, W1, nullptr, ph, DDIM, HID, 512);
    combine_bias_relu_kernel<<<(BDIM * HID + 255) / 256, 256, 0, stream>>>(ph, b1, h);

    // GEMM2 (direct, bias fused): e = h @ W2 + b2  (M=64,K=1024,N=50000; 1563 strips)
    wmma_gemm_strip<true, false, false>
        <<<dim3((NCLS + 31) / 32, 1), wave, 0, stream>>>(h, W2, b2, e, HID, NCLS, HID);

    rowmax_kernel<<<BDIM, 256, 0, stream>>>(e, rmax);
    exp_kernel<<<(BDIM * NCLS + 255) / 256, 256, 0, stream>>>(e, rmax);

    // GEMM3 (split-K x8, tail-guarded): pR[p] = e @ Y_train over K chunk of 6272
    wmma_gemm_strip<false, false, true>
        <<<dim3(DDIM / 32, 8), wave, 0, stream>>>(e, Yt, nullptr, pR, NCLS, DDIM, 6272);
    combine_R_kernel<<<(BDIM * DDIM + 255) / 256, 256, 0, stream>>>(pR, R);

    masked_norm_kernel<<<BDIM, 256, 0, stream>>>(R, x, y);
}
